// MultiModal_19009525252785
// MI455X (gfx1250) — matrix operand order repaired and checked by reading, not compiled, not run on hardware
//
#include <hip/hip_runtime.h>
#include <math.h>

#define HID   768
#define FF    3072
#define NHEADS 12
#define HDIM  64
#define SEQ   512
#define BATCH 16
#define MTOK  (BATCH*SEQ)
#define SCALE 0.125f          // 1/sqrt(64)
#define LDA   40              // LDS row stride (elements): 80B -> conflict-free b128 frags

typedef __attribute__((ext_vector_type(16))) __bf16 v16bf;
typedef __attribute__((ext_vector_type(8)))  __bf16 v8bf;
typedef __attribute__((ext_vector_type(4)))  __bf16 v4bf;
typedef __attribute__((ext_vector_type(8)))  float  v8f;
typedef __attribute__((ext_vector_type(4)))  float  v4f;

union FragAB { v16bf v; v8bf h[2]; };

__device__ __forceinline__ v8f wmma_bf16(const FragAB& a, const FragAB& b, v8f c) {
  // D = A(16x32 bf16) * B(32x16 bf16) + C(16x16 f32)
  return __builtin_amdgcn_wmma_f32_16x16x32_bf16(false, a.v, false, b.v, (short)0, c, false, false);
}

// ---------------------------------------------------------------- cvt f32->bf16
__global__ __launch_bounds__(256) void cvt_f32_bf16(const float* __restrict__ in,
                                                    __bf16* __restrict__ out, size_t n) {
  size_t i = (size_t)blockIdx.x * blockDim.x + threadIdx.x;
  size_t stride = (size_t)gridDim.x * blockDim.x;
  for (; i < n; i += stride) out[i] = (__bf16)in[i];
}

// ---------------------------------------------------------------- GEMM (bf16 WMMA)
// C[M,N] = A[M,K](bf16) @ W[K,N](f32, converted while staging) + bias
// (optional exact GELU), out fp32 and/or bf16.
// Block: 256 thr (8 waves), tile 128x64, BK=32, double-buffered LDS;
// wave tile 32x32 (2x2 WMMA).
__global__ __launch_bounds__(256) void gemm_bf16_kernel(
    const __bf16* __restrict__ A, const float* __restrict__ Wf,
    const float* __restrict__ bias, float* __restrict__ outF,
    __bf16* __restrict__ outB, int M, int N, int K, int doGelu) {
  __shared__ __bf16 As[2][128 * LDA];   // A tile [128][32] padded, double buffered
  __shared__ __bf16 Bt[2][64 * LDA];    // W tile transposed [n][k], double buffered
  const int tid  = threadIdx.x;
  const int bm   = blockIdx.y * 128, bn = blockIdx.x * 64;
  const int wave = tid >> 5, lane = tid & 31, lr = lane & 15, half = lane >> 4;
  const int wm   = wave >> 1, wn = wave & 1;
  v8f acc[2][2] = {};

  auto stage = [&](int buf, int k0) {
#pragma unroll
    for (int i = 0; i < 2; ++i) {                       // A: 128x32, 8-elt chunks
      int ch = tid + i * 256;
      int r = ch >> 2, c = (ch & 3) * 8;
      *(v8bf*)&As[buf][r * LDA + c] = *(const v8bf*)(A + (size_t)(bm + r) * K + k0 + c);
    }
    {                                                   // W: 32x64 f32 -> bf16, transposed
      int kr = tid >> 3, c = (tid & 7) * 8;
      const float* wp = Wf + (size_t)(k0 + kr) * N + bn + c;
      v4f w0 = *(const v4f*)wp;
      v4f w1 = *(const v4f*)(wp + 4);
#pragma unroll
      for (int j = 0; j < 4; ++j) Bt[buf][(c + j) * LDA + kr]     = (__bf16)w0[j];
#pragma unroll
      for (int j = 0; j < 4; ++j) Bt[buf][(c + 4 + j) * LDA + kr] = (__bf16)w1[j];
    }
  };

  stage(0, 0);
  int buf = 0;
  for (int k0 = 0; k0 < K; k0 += 32, buf ^= 1) {
    __syncthreads();                                    // stage(buf) visible; prev reads done
    if (k0 + 32 < K) stage(buf ^ 1, k0 + 32);           // overlap next stage with compute
    if (k0 + 64 < K)                                    // prefetch A two tiles ahead into L2
      __builtin_prefetch(A + (size_t)(bm + (tid >> 2)) * K + k0 + 64, 0, 1);

#pragma unroll
    for (int tm = 0; tm < 2; ++tm) {
      FragAB a;                                          // ISA 16-bit A layout
      const __bf16* ap = &As[buf][(wm * 32 + tm * 16 + lr) * LDA];
      a.h[0] = *(const v8bf*)(ap + half * 8);
      a.h[1] = *(const v8bf*)(ap + 16 + half * 8);
#pragma unroll
      for (int tn = 0; tn < 2; ++tn) {
        FragAB b;                                        // ISA 16-bit B layout
        const __bf16* bp = &Bt[buf][(wn * 32 + tn * 16 + lr) * LDA];
        b.h[0] = *(const v8bf*)(bp + half * 8);
        b.h[1] = *(const v8bf*)(bp + 16 + half * 8);
        acc[tm][tn] = wmma_bf16(a, b, acc[tm][tn]);
      }
    }
  }

#pragma unroll
  for (int tm = 0; tm < 2; ++tm)
#pragma unroll
    for (int tn = 0; tn < 2; ++tn) {
      int col = bn + wn * 32 + tn * 16 + lr;
      float bv = bias ? bias[col] : 0.f;
#pragma unroll
      for (int r = 0; r < 8; ++r) {                      // C layout: row = r + 8*half
        int row = bm + wm * 32 + tm * 16 + half * 8 + r;
        float x = acc[tm][tn][r] + bv;
        if (doGelu) x = 0.5f * x * (1.f + erff(x * 0.70710678118f));
        size_t idx = (size_t)row * N + col;
        if (outF) outF[idx] = x;
        if (outB) outB[idx] = (__bf16)x;
      }
    }
}

// ---------------------------------------------------------------- flash attention
// Block = (64 q rows, 1 head, 1 batch), 4 waves x 16 q rows, k-tiles of 32 keys.
__global__ __launch_bounds__(128) void flash_attn_kernel(
    const __bf16* __restrict__ Qg, const __bf16* __restrict__ Kg,
    const __bf16* __restrict__ Vg, const float* __restrict__ mask,
    __bf16* __restrict__ Og) {
  const int qblk = blockIdx.x, h = blockIdx.y, b = blockIdx.z;
  const int tid = threadIdx.x;
  const int wave = tid >> 5, lane = tid & 31, lr = lane & 15, half = lane >> 4;

  __shared__ __bf16 VT[HDIM][LDA];        // V tile transposed [hd][key]
  __shared__ __bf16 Pt[4][16][LDA];       // per-wave P tile (C-layout -> A-layout bounce)

  const int qrow = qblk * 64 + wave * 16 + lr;
  const __bf16* qp = Qg + (size_t)(b * SEQ + qrow) * HID + h * HDIM;
  FragAB a0, a1;                           // Q 16x64 as two A fragments (K=0..31, 32..63)
  a0.h[0] = *(const v8bf*)(qp + half * 8);
  a0.h[1] = *(const v8bf*)(qp + 16 + half * 8);
  a1.h[0] = *(const v8bf*)(qp + 32 + half * 8);
  a1.h[1] = *(const v8bf*)(qp + 48 + half * 8);

  v8f O[4] = {};
  float mrow[8], lrow[8];
#pragma unroll
  for (int r = 0; r < 8; ++r) { mrow[r] = -3.0e38f; lrow[r] = 0.f; }

  for (int kb = 0; kb < SEQ; kb += 32) {
    __syncthreads();
#pragma unroll
    for (int i = 0; i < 4; ++i) {         // stage V (32 keys x 64 hd) transposed
      int e = tid + i * 128;
      int key = e >> 4, hd4 = (e & 15) * 4;
      v4bf vv = *(const v4bf*)(Vg + (size_t)(b * SEQ + kb + key) * HID + h * HDIM + hd4);
#pragma unroll
      for (int j = 0; j < 4; ++j) VT[hd4 + j][key] = vv[j];
    }
    __syncthreads();

    // scores S = Q @ K^T : two 16x16 tiles (keys kb+lr and kb+16+lr per lane column)
    v8f s0 = {}, s1 = {};
    {
      const __bf16* kp = Kg + (size_t)(b * SEQ + kb + lr) * HID + h * HDIM;
      FragAB bk0, bk1;
      bk0.h[0] = *(const v8bf*)(kp + half * 8);
      bk0.h[1] = *(const v8bf*)(kp + 16 + half * 8);
      bk1.h[0] = *(const v8bf*)(kp + 32 + half * 8);
      bk1.h[1] = *(const v8bf*)(kp + 48 + half * 8);
      s0 = wmma_bf16(a0, bk0, s0);
      s0 = wmma_bf16(a1, bk1, s0);
    }
    {
      const __bf16* kp = Kg + (size_t)(b * SEQ + kb + 16 + lr) * HID + h * HDIM;
      FragAB bk0, bk1;
      bk0.h[0] = *(const v8bf*)(kp + half * 8);
      bk0.h[1] = *(const v8bf*)(kp + 16 + half * 8);
      bk1.h[0] = *(const v8bf*)(kp + 32 + half * 8);
      bk1.h[1] = *(const v8bf*)(kp + 48 + half * 8);
      s1 = wmma_bf16(a0, bk0, s1);
      s1 = wmma_bf16(a1, bk1, s1);
    }
    float mk0 = 0.f, mk1 = 0.f;
    if (mask) { mk0 = mask[b * SEQ + kb + lr]; mk1 = mask[b * SEQ + kb + 16 + lr]; }

#pragma unroll
    for (int r = 0; r < 8; ++r) {          // online softmax per C-layout row
      float e0 = s0[r] * SCALE + mk0;
      float e1 = s1[r] * SCALE + mk1;
      float mx = fmaxf(e0, e1);
#pragma unroll
      for (int off = 1; off < 16; off <<= 1) mx = fmaxf(mx, __shfl_xor(mx, off, 32));
      float nm = fmaxf(mrow[r], mx);
      float alpha = __expf(mrow[r] - nm);
      float p0 = __expf(e0 - nm), p1 = __expf(e1 - nm);
      float ps = p0 + p1;
#pragma unroll
      for (int off = 1; off < 16; off <<= 1) ps += __shfl_xor(ps, off, 32);
      lrow[r] = lrow[r] * alpha + ps;
      mrow[r] = nm;
#pragma unroll
      for (int n4 = 0; n4 < 4; ++n4) O[n4][r] *= alpha;
      Pt[wave][half * 8 + r][lr]      = (__bf16)p0;   // C-layout scatter into LDS
      Pt[wave][half * 8 + r][16 + lr] = (__bf16)p1;
    }
    asm volatile("s_wait_dscnt 0" ::: "memory");       // own-wave LDS RAW fence

    FragAB pf;                             // reload P in A layout
    pf.h[0] = *(const v8bf*)&Pt[wave][lr][half * 8];
    pf.h[1] = *(const v8bf*)&Pt[wave][lr][16 + half * 8];
#pragma unroll
    for (int n4 = 0; n4 < 4; ++n4) {       // O += P @ V
      FragAB vf;
      const __bf16* vp = &VT[n4 * 16 + lr][0];
      vf.h[0] = *(const v8bf*)(vp + half * 8);
      vf.h[1] = *(const v8bf*)(vp + 16 + half * 8);
      O[n4] = wmma_bf16(pf, vf, O[n4]);
    }
  }

#pragma unroll
  for (int r = 0; r < 8; ++r) {
    float inv = 1.f / lrow[r];
    int row = qblk * 64 + wave * 16 + half * 8 + r;
    __bf16* op = Og + (size_t)(b * SEQ + row) * HID + h * HDIM;
#pragma unroll
    for (int n4 = 0; n4 < 4; ++n4) op[n4 * 16 + lr] = (__bf16)(O[n4][r] * inv);
  }
}

// ---------------------------------------------------------------- LayerNorm(x [+ res])
__global__ __launch_bounds__(256) void ln_res_kernel(
    const float* __restrict__ x, const float* __restrict__ res,
    const float* __restrict__ g, const float* __restrict__ beta,
    float* __restrict__ outF, __bf16* __restrict__ outB) {
  __shared__ float red[256];
  const int row = blockIdx.x, t = threadIdx.x;
  const float* xp = x + (size_t)row * HID;
  float v[3]; float s = 0.f;
#pragma unroll
  for (int i = 0; i < 3; ++i) {
    int j = t + i * 256;
    float val = xp[j];
    if (res) val += res[(size_t)row * HID + j];
    v[i] = val; s += val;
  }
  red[t] = s; __syncthreads();
  for (int o = 128; o > 0; o >>= 1) { if (t < o) red[t] += red[t + o]; __syncthreads(); }
  float mean = red[0] * (1.f / HID);
  __syncthreads();
  float s2 = 0.f;
#pragma unroll
  for (int i = 0; i < 3; ++i) { float d = v[i] - mean; s2 += d * d; }
  red[t] = s2; __syncthreads();
  for (int o = 128; o > 0; o >>= 1) { if (t < o) red[t] += red[t + o]; __syncthreads(); }
  float rstd = rsqrtf(red[0] * (1.f / HID) + 1e-12f);
#pragma unroll
  for (int i = 0; i < 3; ++i) {
    int j = t + i * 256;
    float y = (v[i] - mean) * rstd * g[j] + beta[j];
    size_t idx = (size_t)row * HID + j;
    if (outF) outF[idx] = y;
    if (outB) outB[idx] = (__bf16)y;
  }
}

// ---------------------------------------------------------------- embeddings + LN
__global__ __launch_bounds__(256) void embed_ln_kernel(
    const int* __restrict__ ids, const int* __restrict__ tts,
    const float* __restrict__ wemb, const float* __restrict__ pemb,
    const float* __restrict__ temb, const float* __restrict__ g,
    const float* __restrict__ beta, float* __restrict__ outF, __bf16* __restrict__ outB) {
  __shared__ float red[256];
  const int row = blockIdx.x, t = threadIdx.x;
  const int spos = row % SEQ;
  const int id = ids[row], tt = tts[row];
  float v[3]; float s = 0.f;
#pragma unroll
  for (int i = 0; i < 3; ++i) {
    int j = t + i * 256;
    float val = wemb[(size_t)id * HID + j] + pemb[(size_t)spos * HID + j] +
                temb[(size_t)tt * HID + j];
    v[i] = val; s += val;
  }
  red[t] = s; __syncthreads();
  for (int o = 128; o > 0; o >>= 1) { if (t < o) red[t] += red[t + o]; __syncthreads(); }
  float mean = red[0] * (1.f / HID);
  __syncthreads();
  float s2 = 0.f;
#pragma unroll
  for (int i = 0; i < 3; ++i) { float d = v[i] - mean; s2 += d * d; }
  red[t] = s2; __syncthreads();
  for (int o = 128; o > 0; o >>= 1) { if (t < o) red[t] += red[t + o]; __syncthreads(); }
  float rstd = rsqrtf(red[0] * (1.f / HID) + 1e-12f);
#pragma unroll
  for (int i = 0; i < 3; ++i) {
    int j = t + i * 256;
    float y = (v[i] - mean) * rstd * g[j] + beta[j];
    size_t idx = (size_t)row * HID + j;
    outF[idx] = y;
    outB[idx] = (__bf16)y;
  }
}

// ---------------------------------------------------------------- host orchestration
extern "C" void kernel_launch(void* const* d_in, const int* in_sizes, int n_in,
                              void* d_out, int out_size, void* d_ws, size_t ws_size,
                              hipStream_t stream) {
  (void)in_sizes; (void)n_in; (void)out_size; (void)ws_size;
  const int*   ids  = (const int*)d_in[0];
  const int*   tts  = (const int*)d_in[1];
  const float* vis  = (const float*)d_in[2];
  const float* mask = (const float*)d_in[3];
  auto F = [&](int i) { return (const float*)d_in[i]; };

  char* base = (char*)d_ws;
  size_t off = 0;
  auto carve = [&](size_t bytes) -> void* {
    void* p = base + off;
    off += (bytes + 255) & ~(size_t)255;
    return p;
  };
  const size_t MH = (size_t)MTOK * HID;
  float*  t0      = (float*)carve(MH * 4);
  float*  t1      = (float*)carve(MH * 4);
  __bf16* xbf     = (__bf16*)carve(MH * 2);
  __bf16* qbf     = (__bf16*)carve(MH * 2);
  __bf16* kbf     = (__bf16*)carve(MH * 2);
  __bf16* vbf     = (__bf16*)carve(MH * 2);
  __bf16* ctxbf   = (__bf16*)carve(MH * 2);
  __bf16* hvisbf  = (__bf16*)carve(MH * 2);
  __bf16* hlangbf = (__bf16*)carve(MH * 2);
  __bf16* interbf = (__bf16*)carve((size_t)MTOK * FF * 2);

  auto gemm = [&](const __bf16* A, int wi, float* oF, __bf16* oB, int N, int K, int gelu) {
    gemm_bf16_kernel<<<dim3(N / 64, MTOK / 128), 256, 0, stream>>>(
        A, F(wi), F(wi + 1), oF, oB, MTOK, N, K, gelu);
  };
  const dim3 fgrid(SEQ / 64, NHEADS, BATCH);

  // ---- vision self layer (weights 25..40) ----
  cvt_f32_bf16<<<2048, 256, 0, stream>>>(vis, xbf, MH);
  gemm(xbf, 25, nullptr, qbf, HID, HID, 0);
  gemm(xbf, 27, nullptr, kbf, HID, HID, 0);
  gemm(xbf, 29, nullptr, vbf, HID, HID, 0);
  flash_attn_kernel<<<fgrid, 128, 0, stream>>>(qbf, kbf, vbf, nullptr, ctxbf);
  gemm(ctxbf, 31, t0, nullptr, HID, HID, 0);
  ln_res_kernel<<<MTOK, 256, 0, stream>>>(t0, vis, F(33), F(34), t1, xbf);
  gemm(xbf, 35, nullptr, interbf, FF, HID, 1);
  gemm(interbf, 37, t0, nullptr, HID, FF, 0);
  ln_res_kernel<<<MTOK, 256, 0, stream>>>(t0, t1, F(39), F(40), nullptr, hvisbf);

  // ---- language self layer (weights 9..24) ----
  embed_ln_kernel<<<MTOK, 256, 0, stream>>>(ids, tts, F(4), F(5), F(6), F(7), F(8), t1, xbf);
  gemm(xbf, 9,  nullptr, qbf, HID, HID, 0);
  gemm(xbf, 11, nullptr, kbf, HID, HID, 0);
  gemm(xbf, 13, nullptr, vbf, HID, HID, 0);
  flash_attn_kernel<<<fgrid, 128, 0, stream>>>(qbf, kbf, vbf, mask, ctxbf);
  gemm(ctxbf, 15, t0, nullptr, HID, HID, 0);
  ln_res_kernel<<<MTOK, 256, 0, stream>>>(t0, t1, F(17), F(18), t1, xbf);
  gemm(xbf, 19, nullptr, interbf, FF, HID, 1);
  gemm(interbf, 21, t0, nullptr, HID, FF, 0);
  ln_res_kernel<<<MTOK, 256, 0, stream>>>(t0, t1, F(23), F(24), t1, hlangbf);

  // ---- mixed cross-attention layer (weights 41..50) ----
  gemm(hlangbf, 41, nullptr, qbf, HID, HID, 0);
  gemm(hvisbf,  43, nullptr, kbf, HID, HID, 0);
  gemm(hvisbf,  45, nullptr, vbf, HID, HID, 0);
  flash_attn_kernel<<<fgrid, 128, 0, stream>>>(qbf, kbf, vbf, nullptr, ctxbf);
  gemm(ctxbf, 47, t0, nullptr, HID, HID, 0);
  ln_res_kernel<<<MTOK, 256, 0, stream>>>(t0, t1, F(49), F(50), (float*)d_out, nullptr);
}
